// Attention_30408368456253
// MI455X (gfx1250) — matrix operand order repaired and checked by reading, not compiled, not run on hardware
//
#include <hip/hip_runtime.h>

// Shapes from the reference
#define B_  4
#define N_  4
#define L_  1024
#define H_  8
#define E_  64
#define RS  (H_ * E_)          // 512 floats: row stride of [.., l, h, e] layout

#define TILE_M  128            // query rows per workgroup (8 waves x 16)
#define WAVE_M  16
#define CHUNK   64             // keys per main-loop iteration (2x WMMA K)
#define NWAVES  8

#define KPITCH  72             // ushort pitch, ldsK rows  [key][e]   (conflict-free b128 reads)
#define VPITCH  72             // ushort pitch, ldsVT rows [e][key]   (conflict-free b128 reads)
#define PPITCH  72             // ushort pitch, ldsP rows  [row][key]

// softmax in base-2, folded into Q: scale = (1/sqrt(64)) * log2(e)
#define SCALE2  0.18033688011112042f

typedef __attribute__((ext_vector_type(16))) __bf16          v16bf;
typedef __attribute__((ext_vector_type(16))) unsigned short  v16u;
typedef __attribute__((ext_vector_type(8)))  unsigned short  v8u;
typedef __attribute__((ext_vector_type(8)))  float           v8f;

__device__ __forceinline__ unsigned short f2bf(float f) {
    return __builtin_bit_cast(unsigned short, (__bf16)f);   // native v_cvt
}

// 16-lane max reduction with fused-DPP max (1 instr/step)
__device__ __forceinline__ float rowmax16(float x) {
    float t;
    asm("v_max_num_f32_dpp %0, %1, %1 quad_perm:[1,0,3,2] row_mask:0xf bank_mask:0xf bound_ctrl:1\n\t"
        "v_max_num_f32_dpp %0, %0, %0 quad_perm:[2,3,0,1] row_mask:0xf bank_mask:0xf bound_ctrl:1\n\t"
        "v_max_num_f32_dpp %0, %0, %0 row_half_mirror row_mask:0xf bank_mask:0xf bound_ctrl:1\n\t"
        "v_max_num_f32_dpp %0, %0, %0 row_mirror row_mask:0xf bank_mask:0xf bound_ctrl:1"
        : "=&v"(t) : "v"(x));
    return t;
}

// 16 bf16 from two aligned 16B runs (8 at lo -> elems 0-7, 8 at hi -> 8-15) -> fragment register
__device__ __forceinline__ v16bf frag_from_lds(const unsigned short* lo,
                                               const unsigned short* hi) {
    v8u a = *(const v8u*)lo;
    v8u b = *(const v8u*)hi;
    v16u t = __builtin_shufflevector(a, b, 0,1,2,3,4,5,6,7,8,9,10,11,12,13,14,15);
    return __builtin_bit_cast(v16bf, t);
}

#define WMMA_BF16(A, B, C) \
    __builtin_amdgcn_wmma_f32_16x16x32_bf16(false, (A), false, (B), (short)0, (C), false, false)

__global__ __launch_bounds__(256)
void fa_causal_kernel(const float* __restrict__ Q,
                      const float* __restrict__ K,
                      const float* __restrict__ V,
                      float* __restrict__ O) {
    __shared__ alignas(16) unsigned short ldsK[CHUNK * KPITCH];           // [key][e]
    __shared__ alignas(16) unsigned short ldsVT[E_ * VPITCH];             // [e][key] transposed
    __shared__ alignas(16) unsigned short ldsP[NWAVES][WAVE_M * PPITCH];  // per-wave P tile

    const int tid   = threadIdx.x;
    const int lane  = tid & 31;
    const int wid   = tid >> 5;          // 0..7
    const int n     = lane & 15;
    const int hhalf = lane >> 4;
    const int khalf = hhalf * 8;

    const int head = blockIdx.x >> 3;    // 0..127
    const int tile = blockIdx.x & 7;     // 0..7
    const int bn   = head >> 3;
    const int h    = head & 7;
    const size_t base = (size_t)bn * L_ * RS + (size_t)h * E_;
    const float* Qb = Q + base;
    const float* Kb = K + base;
    const float* Vb = V + base;
    float*       Ob = O + base;

    const int row0 = tile * TILE_M + wid * WAVE_M;

    // staging coords: 256 threads cover a 64x64 chunk, 16 contiguous floats each
    const int srow = tid >> 2;           // 0..63  (key row)
    const int scol = (tid & 3) * 16;     // 0,16,32,48 (e column base)

    // all-ones bf16 B fragment: C[m,n] = row-sum of A -> softmax denominator via WMMA
    v16bf bOnes;
    {
        v16u o;
        #pragma unroll
        for (int i = 0; i < 16; ++i) o[i] = 0x3F80;   // bf16 1.0
        bOnes = __builtin_bit_cast(v16bf, o);
    }

    // ---- Q A-fragments (loaded once, pre-scaled by 0.125*log2e so scores
    //      come out in base-2 softmax units with no per-chunk multiply) ----
    v16bf aQ[2];
    {
        const float* qp = Qb + (size_t)(row0 + n) * RS;
        #pragma unroll
        for (int f = 0; f < 2; ++f) {
            v16u t;
            #pragma unroll
            for (int i = 0; i < 8; ++i) t[i]     = f2bf(qp[f * 32 + khalf + i] * SCALE2);
            #pragma unroll
            for (int i = 0; i < 8; ++i) t[8 + i] = f2bf(qp[f * 32 + 16 + khalf + i] * SCALE2);
            aQ[f] = __builtin_bit_cast(v16bf, t);
        }
    }

    float mrow[8];
    v8f acc[4], accL;
    #pragma unroll
    for (int r = 0; r < 8; ++r) mrow[r] = -1e30f;
    #pragma unroll
    for (int t = 0; t < 4; ++t) acc[t] = v8f{0.f,0.f,0.f,0.f,0.f,0.f,0.f,0.f};
    accL = v8f{0.f,0.f,0.f,0.f,0.f,0.f,0.f,0.f};

    const int jmax = tile * TILE_M + TILE_M;   // causal bound (uniform per WG)

    for (int jj = 0; jj < jmax; jj += CHUNK) {
        __syncthreads();   // previous chunk's LDS fully consumed

        // ---- cooperative staging: K (row-major bf16) and V (transposed bf16) ----
        {
            const float4* kp = (const float4*)(Kb + (size_t)(jj + srow) * RS + scol);
            float4 kx[4];
            #pragma unroll
            for (int q = 0; q < 4; ++q) kx[q] = kp[q];
            #pragma unroll
            for (int q = 0; q < 2; ++q) {
                v8u w;
                w[0]=f2bf(kx[2*q].x);   w[1]=f2bf(kx[2*q].y);
                w[2]=f2bf(kx[2*q].z);   w[3]=f2bf(kx[2*q].w);
                w[4]=f2bf(kx[2*q+1].x); w[5]=f2bf(kx[2*q+1].y);
                w[6]=f2bf(kx[2*q+1].z); w[7]=f2bf(kx[2*q+1].w);
                *(v8u*)&ldsK[srow * KPITCH + scol + q * 8] = w;
            }

            const float4* vp = (const float4*)(Vb + (size_t)(jj + srow) * RS + scol);
            float4 vx[4];
            #pragma unroll
            for (int q = 0; q < 4; ++q) vx[q] = vp[q];
            #pragma unroll
            for (int q = 0; q < 4; ++q) {
                ldsVT[(scol + q*4 + 0) * VPITCH + srow] = f2bf(vx[q].x);
                ldsVT[(scol + q*4 + 1) * VPITCH + srow] = f2bf(vx[q].y);
                ldsVT[(scol + q*4 + 2) * VPITCH + srow] = f2bf(vx[q].z);
                ldsVT[(scol + q*4 + 3) * VPITCH + srow] = f2bf(vx[q].w);
            }

            if (jj + CHUNK < jmax) {   // prefetch next chunk into cache
                __builtin_prefetch(Kb + (size_t)(jj + CHUNK + srow) * RS + scol, 0, 3);
                __builtin_prefetch(Vb + (size_t)(jj + CHUNK + srow) * RS + scol, 0, 3);
            }
        }
        __syncthreads();

        // ---- scores (already in base-2 softmax units): four 16x16 tiles ----
        v8f sc[4];
        #pragma unroll
        for (int s = 0; s < 4; ++s) {
            const unsigned short* kr = &ldsK[(s * 16 + n) * KPITCH + khalf];
            v16bf b0 = frag_from_lds(kr,      kr + 16);
            v16bf b1 = frag_from_lds(kr + 32, kr + 48);
            v8f z = v8f{0.f,0.f,0.f,0.f,0.f,0.f,0.f,0.f};
            z = WMMA_BF16(aQ[0], b0, z);
            sc[s] = WMMA_BF16(aQ[1], b1, z);
        }

        // ---- mask (only on diagonal chunk) + online softmax (exp2) ----
        const bool domask = (jj + CHUNK - 1) > row0;   // wave-uniform
        float corr[8];
        unsigned short* pw = &ldsP[wid][0];
        #pragma unroll
        for (int r = 0; r < 8; ++r) {
            const int qrow = row0 + r + khalf;         // C-layout: M = r + 8*hhalf
            float v0 = sc[0][r];
            float v1 = sc[1][r];
            float v2 = sc[2][r];
            float v3 = sc[3][r];
            if (domask) {
                if (jj + n      > qrow) v0 = -1e30f;
                if (jj + 16 + n > qrow) v1 = -1e30f;
                if (jj + 32 + n > qrow) v2 = -1e30f;
                if (jj + 48 + n > qrow) v3 = -1e30f;
            }
            const float mx   = rowmax16(fmaxf(fmaxf(v0, v1), fmaxf(v2, v3)));
            const float mnew = fmaxf(mrow[r], mx);
            corr[r] = __builtin_amdgcn_exp2f(mrow[r] - mnew);
            mrow[r] = mnew;

            v0 = __builtin_amdgcn_exp2f(v0 - mnew);
            v1 = __builtin_amdgcn_exp2f(v1 - mnew);
            v2 = __builtin_amdgcn_exp2f(v2 - mnew);
            v3 = __builtin_amdgcn_exp2f(v3 - mnew);
            const int ro = (r + khalf) * PPITCH;
            pw[ro + n]      = f2bf(v0);
            pw[ro + 16 + n] = f2bf(v1);
            pw[ro + 32 + n] = f2bf(v2);
            pw[ro + 48 + n] = f2bf(v3);
        }

        // rescale accumulators (output + denominator) by correction factors
        #pragma unroll
        for (int r = 0; r < 8; ++r) {
            const float cr = corr[r];
            accL[r]   *= cr;
            acc[0][r] *= cr;
            acc[1][r] *= cr;
            acc[2][r] *= cr;
            acc[3][r] *= cr;
        }

        // ---- O += P*V and L += P*1 : two 32-key A-fragments ----
        #pragma unroll
        for (int g = 0; g < 2; ++g) {
            const unsigned short* pr = &ldsP[wid][n * PPITCH + g * 32];
            const v16bf aP = frag_from_lds(pr + khalf, pr + 16 + khalf);
            accL = WMMA_BF16(aP, bOnes, accL);
            #pragma unroll
            for (int t = 0; t < 4; ++t) {
                const unsigned short* vr = &ldsVT[(t * 16 + n) * VPITCH + g * 32 + khalf];
                const v16bf bV = frag_from_lds(vr, vr + 16);
                acc[t] = WMMA_BF16(aP, bV, acc[t]);
            }
        }
    }

    // ---- normalize and store (accL holds the row sum in every column) ----
    #pragma unroll
    for (int r = 0; r < 8; ++r) {
        const float inv = __builtin_amdgcn_rcpf(accL[r]);
        const size_t ro = (size_t)(row0 + r + khalf) * RS;
        #pragma unroll
        for (int t = 0; t < 4; ++t)
            Ob[ro + t * 16 + n] = acc[t][r] * inv;
    }
}

extern "C" void kernel_launch(void* const* d_in, const int* in_sizes, int n_in,
                              void* d_out, int out_size, void* d_ws, size_t ws_size,
                              hipStream_t stream) {
    (void)in_sizes; (void)n_in; (void)out_size; (void)d_ws; (void)ws_size;
    const float* Q = (const float*)d_in[0];
    const float* K = (const float*)d_in[1];
    const float* V = (const float*)d_in[2];
    float*       O = (float*)d_out;

    dim3 grid(B_ * N_ * H_ * (L_ / TILE_M));   // 128 heads * 8 row tiles = 1024
    dim3 block(256);                            // 8 waves (wave32)
    hipLaunchKernelGGL(fa_causal_kernel, grid, block, 0, stream, Q, K, V, O);
}
